// ARNN_bin_71107478552689
// MI455X (gfx1250) — matrix operand order repaired and checked by reading, not compiled, not run on hardware
//
#include <hip/hip_runtime.h>

// ---------------- problem constants ----------------
#define Bdim 128
#define Tdim 64
#define Idim 512
#define Hdim 1024
#define Odim 512
#define NMAX 12
#define EPSC 0.01f

typedef __attribute__((ext_vector_type(8)))  __bf16 v8bf;
typedef __attribute__((ext_vector_type(16))) __bf16 v16bf;
typedef __attribute__((ext_vector_type(8)))  float  v8f;

// ---------------- WMMA helpers (CDNA5 wave32, 16x16x32 bf16) ----------------
static __device__ inline v16bf cat8(v8bf a, v8bf b) {
  v16bf r;
#pragma unroll
  for (int e = 0; e < 8; ++e) { r[e] = a[e]; r[8 + e] = b[e]; }
  return r;
}

// A matrix (M=16 x K=32 slice) from row-major [m][k], row stride ld elements.
// lane = hi*16+l -> m=l ; elems e=0..15: k=kb+hi*16+e (contiguous, matches load_frag_b)
static __device__ inline v16bf load_frag_a(const __bf16* base, int ld, int kb, int lane) {
  int l = lane & 15, hi = lane >> 4;
  const __bf16* p = base + l * ld + kb;
  v8bf c0 = *(const v8bf*)(p + hi * 16);
  v8bf c1 = *(const v8bf*)(p + hi * 16 + 8);
  return cat8(c0, c1);
}

// B matrix (K=32 x N=16 slice) where B[k][n] = W[n][k], W row-major with stride ld.
// lane -> n=l ; elems e=0..15: k = kb + hi*16 + e  (contiguous in W row n)
static __device__ inline v16bf load_frag_b(const __bf16* base, int ld, int kb, int lane) {
  int l = lane & 15, hi = lane >> 4;
  const __bf16* p = base + l * ld + kb + hi * 16;
  v8bf c0 = *(const v8bf*)(p);
  v8bf c1 = *(const v8bf*)(p + 8);
  return cat8(c0, c1);
}

static __device__ inline v8f wmma_bf16(v16bf a, v16bf b, v8f c) {
  return __builtin_amdgcn_wmma_f32_16x16x32_bf16(false, a, false, b, (short)0, c,
                                                 false, false);
}

// ---------------- conversion kernels ----------------
__global__ void k_f2bf(const float* __restrict__ s, __bf16* __restrict__ d, int n) {
  int i = blockIdx.x * blockDim.x + threadIdx.x;
  if (i < n) d[i] = (__bf16)s[i];
}

// w_ih is (H, 1+I): split column 0 (the "ones" bias column) from the rest.
__global__ void k_conv_wih(const float* __restrict__ w_ih, __bf16* __restrict__ wih_bf,
                           float* __restrict__ wih0) {
  int i = blockIdx.x * blockDim.x + threadIdx.x;
  if (i < Hdim * Idim) {
    int h = i >> 9, c = i & (Idim - 1);
    wih_bf[i] = (__bf16)w_ih[h * (Idim + 1) + 1 + c];
  }
  if (i < Hdim) wih0[i] = w_ih[i * (Idim + 1)];
}

// ---------------- XW = x @ w_ih[:,1:].T + b_ih + b_hh ----------------
// rows = B*T = 8192, cols = H = 1024, K = I = 512.  One 16x16 tile per wave.
__global__ __launch_bounds__(256) void k_xw(const __bf16* __restrict__ xb,
                                            const __bf16* __restrict__ wih,
                                            const float* __restrict__ b_ih,
                                            const float* __restrict__ b_hh,
                                            float* __restrict__ XW) {
  int lane = threadIdx.x & 31;
  int gw = blockIdx.x * 8 + (threadIdx.x >> 5);   // 32768 wave-tiles
  int mt = gw >> 6;                                // 512 M tiles
  int nt = gw & 63;                                // 64  N tiles
  const __bf16* abase = xb + (size_t)mt * 16 * Idim;
  const __bf16* bbase = wih + (size_t)nt * 16 * Idim;
  v8f c = {0, 0, 0, 0, 0, 0, 0, 0};
  for (int kb = 0; kb < Idim; kb += 32) {
    v16bf a = load_frag_a(abase, Idim, kb, lane);
    v16bf b = load_frag_b(bbase, Idim, kb, lane);
    c = wmma_bf16(a, b, c);
  }
  int l = lane & 15, hi = lane >> 4;
  int nn = nt * 16 + l;
  float bias = b_ih[nn] + b_hh[nn];
#pragma unroll
  for (int r = 0; r < 8; ++r) {
    int row = mt * 16 + hi * 8 + r;
    XW[(size_t)row * Hdim + nn] = c[r] + bias;
  }
}

// ---------------- persistent ACT-RNN kernel ----------------
// grid = B/16 = 8 workgroups, 1024 threads = 32 waves each (fills one WGP).
// Wave w owns h-columns [w*32, w*32+32) (2 tiles) and out-column tile w (1 tile).
__global__ __launch_bounds__(1024, 1) void k_rnn(const float* __restrict__ XW,
                                                 const __bf16* __restrict__ whh,
                                                 const __bf16* __restrict__ wout,
                                                 const float* __restrict__ wih0,
                                                 const float* __restrict__ w_halt,
                                                 const float* __restrict__ b_halt,
                                                 const float* __restrict__ b_out,
                                                 const float* __restrict__ s0,
                                                 float* __restrict__ out,
                                                 float* __restrict__ pond) {
  __shared__ __attribute__((aligned(16))) __bf16 h_bf[16][Hdim];  // 32 KB
  __shared__ float wih0_s[Hdim];                                  // 4 KB
  __shared__ float whalt_s[Hdim];                                 // 4 KB
  __shared__ float part[16][64];                                  // 4 KB
  __shared__ float cum_s[16];
  __shared__ float act_s[16];
  __shared__ int   cnt_s[16];
  __shared__ int   anyact_s;

  const int tid  = threadIdx.x;
  const int wave = tid >> 5;      // 0..31
  const int lane = tid & 31;
  const int l    = lane & 15;
  const int hi   = lane >> 4;
  const int b0   = blockIdx.x * 16;

  for (int i = tid; i < Hdim; i += 1024) {
    wih0_s[i]  = wih0[i];
    whalt_s[i] = w_halt[i];
  }
  // initial hidden state h_carry = s0  (s0 is (1,B,H))
  for (int i = tid; i < 16 * Hdim; i += 1024) {
    int m = i >> 10, k = i & (Hdim - 1);
    h_bf[m][k] = (__bf16)s0[(size_t)(b0 + m) * Hdim + k];
  }
  __syncthreads();

  const float bh = b_halt[0];

  for (int t = 0; t < Tdim; ++t) {
    v8f accO;
    v8f hacc[2];
#pragma unroll
    for (int r = 0; r < 8; ++r) accO[r] = 0.0f;
#pragma unroll
    for (int j = 0; j < 2; ++j)
#pragma unroll
      for (int r = 0; r < 8; ++r) hacc[j][r] = 0.0f;

    if (tid < 16) { cum_s[tid] = 0.0f; act_s[tid] = 1.0f; cnt_s[tid] = 0; }
    if (tid == 0) anyact_s = 1;

    const float* xw_t = XW + ((size_t)b0 * Tdim + t) * Hdim;
    // warm the 64KB XW slice for this timestep (global_prefetch_b8); the
    // epilogue reads sit on the GEMM1->GEMM2 critical path every inner step.
    __builtin_prefetch(xw_t + (size_t)(tid >> 6) * Tdim * Hdim + (tid & 63) * 16, 0, 3);
    __syncthreads();

    for (int n = 0; n < NMAX; ++n) {
      // ---- GEMM1: pre = h_prev @ w_hh.T  (M=16, N=1024, K=1024) ----
      v8f hpre[2];
#pragma unroll
      for (int j = 0; j < 2; ++j)
#pragma unroll
        for (int r = 0; r < 8; ++r) hpre[j][r] = 0.0f;

      for (int kb = 0; kb < Hdim; kb += 32) {
        v16bf a = load_frag_a(&h_bf[0][0], Hdim, kb, lane);
#pragma unroll
        for (int j = 0; j < 2; ++j) {
          int n0 = (wave * 2 + j) * 16;
          v16bf b = load_frag_b(whh + (size_t)n0 * Hdim, Hdim, kb, lane);
          hpre[j] = wmma_bf16(a, b, hpre[j]);
        }
      }
      __syncthreads();  // everyone done reading old h_bf

      // ---- epilogue: + XW (+ w_ih[:,0] on step 0), tanh, active mask ----
#pragma unroll
      for (int j = 0; j < 2; ++j) {
        int nn = (wave * 2 + j) * 16 + l;
        float w0 = (n == 0) ? wih0_s[nn] : 0.0f;
#pragma unroll
        for (int r = 0; r < 8; ++r) {
          int m = hi * 8 + r;
          float z  = hpre[j][r] + xw_t[(size_t)m * Tdim * Hdim + nn] + w0;
          float hv = tanhf(z) * act_s[m];
          hacc[j][r] += hv;                  // h_acc += h*w (w==act)
          h_bf[m][nn] = (__bf16)hv;
        }
      }
      __syncthreads();  // new h_bf visible

      // ---- GEMM2: out_acc += h @ w_out.T  (M=16, N=512, K=1024) ----
      {
        int n0 = wave * 16;
        for (int kb = 0; kb < Hdim; kb += 32) {
          v16bf a = load_frag_a(&h_bf[0][0], Hdim, kb, lane);
          v16bf b = load_frag_b(wout + (size_t)n0 * Hdim, Hdim, kb, lane);
          accO = wmma_bf16(a, b, accO);
        }
      }

      // ---- halting GEMV: p = sigmoid(h @ w_halt + b_halt) * act ----
      {
        int m = tid >> 6, seg = tid & 63;
        float s = 0.0f;
        const __bf16* hr = &h_bf[m][seg * 16];
#pragma unroll
        for (int k = 0; k < 16; ++k) s += (float)hr[k] * whalt_s[seg * 16 + k];
        part[m][seg] = s;
      }
      __syncthreads();
      if (tid < 16) {
        float s = 0.0f;
#pragma unroll
        for (int k = 0; k < 64; ++k) s += part[tid][k];
        float a = act_s[tid];
        float p = a * (1.0f / (1.0f + __expf(-(s + bh))));
        float c = cum_s[tid] + p;
        cum_s[tid] = c;
        cnt_s[tid] += (a > 0.0f) ? 1 : 0;
        act_s[tid] = (c < 1.0f - EPSC) ? 1.0f : 0.0f;
        if (n == 0)
          pond[(size_t)(b0 + tid) * Tdim + t] = (c >= 1.0f - EPSC) ? 2.0f : 0.0f;
      }
      __syncthreads();
      if (tid == 0) {
        int any = 0;
#pragma unroll
        for (int m = 0; m < 16; ++m) any |= (act_s[m] > 0.0f) ? 1 : 0;
        anyact_s = any;
      }
      __syncthreads();
      if (!anyact_s) break;  // remaining ponder steps contribute exactly zero
    }

    // ---- timestep end: outputs (+ cnt * b_out) and h_carry = h_acc ----
    {
      int oo = wave * 16 + l;
      float bo = b_out[oo];
#pragma unroll
      for (int r = 0; r < 8; ++r) {
        int m = hi * 8 + r;
        out[((size_t)(b0 + m) * Tdim + t) * Odim + oo] =
            accO[r] + (float)cnt_s[m] * bo;
      }
    }
    __syncthreads();
#pragma unroll
    for (int j = 0; j < 2; ++j) {
      int nn = (wave * 2 + j) * 16 + l;
#pragma unroll
      for (int r = 0; r < 8; ++r) {
        int m = hi * 8 + r;
        h_bf[m][nn] = (__bf16)hacc[j][r];
      }
    }
    __syncthreads();
  }
}

// ---------------- workspace layout (bytes) ----------------
#define OFF_XW    ((size_t)0)                         // 8192*1024*4 = 33554432
#define OFF_XBF   ((size_t)33554432)                  // 8192*512*2  =  8388608
#define OFF_WHHB  ((size_t)41943040)                  // 1024*1024*2 =  2097152
#define OFF_WOUTB ((size_t)44040192)                  // 512*1024*2  =  1048576
#define OFF_WIHB  ((size_t)45088768)                  // 1024*512*2  =  1048576
#define OFF_WIH0  ((size_t)46137344)                  // 1024*4

extern "C" void kernel_launch(void* const* d_in, const int* in_sizes, int n_in,
                              void* d_out, int out_size, void* d_ws, size_t ws_size,
                              hipStream_t stream) {
  const float* x      = (const float*)d_in[0];
  const float* s0     = (const float*)d_in[1];
  const float* w_ih   = (const float*)d_in[2];
  const float* w_hh   = (const float*)d_in[3];
  const float* b_ih   = (const float*)d_in[4];
  const float* b_hh   = (const float*)d_in[5];
  const float* w_halt = (const float*)d_in[6];
  const float* b_halt = (const float*)d_in[7];
  const float* w_out  = (const float*)d_in[8];
  const float* b_out  = (const float*)d_in[9];

  char* ws = (char*)d_ws;
  float*  XW    = (float*)(ws + OFF_XW);
  __bf16* xb    = (__bf16*)(ws + OFF_XBF);
  __bf16* whhb  = (__bf16*)(ws + OFF_WHHB);
  __bf16* woutb = (__bf16*)(ws + OFF_WOUTB);
  __bf16* wihb  = (__bf16*)(ws + OFF_WIHB);
  float*  wih0  = (float*)(ws + OFF_WIH0);

  float* outp = (float*)d_out;
  float* pond = outp + (size_t)Bdim * Tdim * Odim;

  const int nx = Bdim * Tdim * Idim;   // 4194304
  k_f2bf<<<(nx + 255) / 256, 256, 0, stream>>>(x, xb, nx);
  k_f2bf<<<(Hdim * Hdim + 255) / 256, 256, 0, stream>>>(w_hh, whhb, Hdim * Hdim);
  k_f2bf<<<(Odim * Hdim + 255) / 256, 256, 0, stream>>>(w_out, woutb, Odim * Hdim);
  k_conv_wih<<<(Hdim * Idim + 255) / 256, 256, 0, stream>>>(w_ih, wihb, wih0);

  // 512 M-tiles * 64 N-tiles = 32768 wave-tiles / 8 waves per block
  k_xw<<<4096, 256, 0, stream>>>(xb, wihb, b_ih, b_hh, XW);

  k_rnn<<<Bdim / 16, 1024, 0, stream>>>(XW, whhb, woutb, wih0, w_halt, b_halt,
                                        b_out, s0, outp, pond);
}
